// GACF_13907104105032
// MI455X (gfx1250) — matrix operand order repaired and checked by reading, not compiled, not run on hardware
//
#include <hip/hip_runtime.h>

// ---------------------------------------------------------------------------
// GACF forward for MI455X (gfx1250, wave32, WMMA).
// Heavy GEMMs -> v_wmma_f32_16x16x32_f16 (f16 operands, f32 accum).
// Attention  -> flash-style streaming softmax (no NxM score materialization).
// Graph prop -> edge-parallel global_atomic_add_f32.
// B operands pre-transposed so all hot LDS traffic is b128.
// Tiles staged with global_load_async_to_lds_b128, double-buffered so the
// ASYNCcnt DMA of tile i+1 overlaps the WMMA compute of tile i.
// ---------------------------------------------------------------------------

typedef _Float16 f16x16 __attribute__((ext_vector_type(16)));
typedef _Float16 f16x8  __attribute__((ext_vector_type(8)));
typedef float    f32x8  __attribute__((ext_vector_type(8)));

#define N_NODES 7000
#define U_NUM   4000
#define I_NUM   3000
#define EMBED   256
#define FINCOLS 448   // 256+128+64

__device__ inline f16x16 frag_cat(f16x8 lo, f16x8 hi) {
  return __builtin_shufflevector(lo, hi, 0, 1, 2, 3, 4, 5, 6, 7, 8, 9, 10, 11, 12, 13, 14, 15);
}
__device__ inline f16x8 lds_ld8(const _Float16* p) {
  return *reinterpret_cast<const f16x8*>(p);
}
__device__ inline f16x8 g_ld8(const _Float16* p, bool ok) {
  f16x8 z = {};
  if (ok) z = *reinterpret_cast<const f16x8*>(p);
  return z;
}
__device__ inline void async_b128(void* lds, const _Float16* gsrc) {
  unsigned loff = (unsigned)(unsigned long long)lds;
  unsigned long long gaddr = (unsigned long long)gsrc;
  asm volatile("global_load_async_to_lds_b128 %0, %1, off" :: "v"(loff), "v"(gaddr) : "memory");
}
__device__ inline void wait_async0() {
  asm volatile("s_wait_asynccnt 0x0" ::: "memory");
}

// ---------------------------------------------------------------------------
// f32 (K x N) -> f16 transposed (N x K)
// ---------------------------------------------------------------------------
__global__ void castT_f32_f16(const float* __restrict__ s, _Float16* __restrict__ d,
                              int K, int N) {
  int i = blockIdx.x * blockDim.x + threadIdx.x;
  if (i < K * N) {
    int k = i / N, n = i % N;
    d[(long)n * K + k] = (_Float16)s[i];
  }
}

// ---------------------------------------------------------------------------
// concat embeddings into final[:,0:256] (f32, ld 448) and feat16 (f16, ld 256)
// ---------------------------------------------------------------------------
__global__ void embed_kernel(const float* __restrict__ uE, const float* __restrict__ iE,
                             float* __restrict__ finalBuf, _Float16* __restrict__ feat16) {
  long idx = (long)blockIdx.x * blockDim.x + threadIdx.x;
  if (idx >= (long)N_NODES * EMBED) return;
  int r = (int)(idx / EMBED), c = (int)(idx % EMBED);
  float v = (r < U_NUM) ? uE[(long)r * EMBED + c] : iE[(long)(r - U_NUM) * EMBED + c];
  finalBuf[(long)r * FINCOLS + c] = v;
  feat16[idx] = (_Float16)v;
}

// ---------------------------------------------------------------------------
// WMMA GEMM: C[MxN](f32) = act(A[MxK](f16) @ B[KxN](f16) + bias)
// B supplied TRANSPOSED: BT is N x K (ldbt = K). Requires K%32==0, N%64==0.
// Block = 256 (8 waves); tile 64x64; double-buffered async LDS staging.
// ---------------------------------------------------------------------------
__global__ __launch_bounds__(256)
void wmma_gemm_kernel(const _Float16* __restrict__ A, int lda,
                      const _Float16* __restrict__ BT, int ldbt,
                      const float* __restrict__ bias,
                      float* __restrict__ C, int ldc,
                      _Float16* __restrict__ Ch, int ldch,
                      int M, int N, int K, int relu) {
  __shared__ _Float16 As[2][64][40];   // [buf][m][k], row stride 80B (16B aligned)
  __shared__ _Float16 Bs[2][64][40];   // [buf][n][k]

  const int tid  = threadIdx.x;
  const int wave = tid >> 5;
  const int lane = tid & 31;
  const int lr   = lane & 15;
  const int hi   = lane >> 4;
  const int m0   = blockIdx.x * 64;
  const int n0   = blockIdx.y * 64;
  const int tr   = wave >> 1;   // 0..3
  const int tc   = wave & 1;    // 0..1

  f32x8 c0 = {}; f32x8 c1 = {};

  const int srow = tid >> 2;          // 0..63 (A row / BT row)
  const int scol = (tid & 3) * 8;     // 0,8,16,24 (k chunk)
  const bool interior = (m0 + 64) <= M;
  const _Float16* aP = A + (long)(m0 + srow) * lda + scol;
  const _Float16* bP = BT + (long)(n0 + srow) * ldbt + scol;

  auto stage = [&](int k0, int buf) {
    if (interior) {
      async_b128(&As[buf][srow][scol], aP + k0);
    } else {
      *reinterpret_cast<f16x8*>(&As[buf][srow][scol]) = g_ld8(aP + k0, (m0 + srow) < M);
    }
    async_b128(&Bs[buf][srow][scol], bP + k0);
  };

  stage(0, 0);
  wait_async0();
  __syncthreads();

  int buf = 0;
  for (int k0 = 0; k0 < K; k0 += 32, buf ^= 1) {
    if (k0 + 32 < K) stage(k0 + 32, buf ^ 1);   // DMA next tile while computing

    const _Float16* ap  = &As[buf][tr * 16 + lr][0];
    const _Float16* b0p = &Bs[buf][tc * 32 + lr][0];
    const _Float16* b1p = &Bs[buf][tc * 32 + 16 + lr][0];
    f16x16 a  = frag_cat(lds_ld8(ap + hi * 16), lds_ld8(ap + hi * 16 + 8));
    f16x16 b0 = frag_cat(lds_ld8(b0p + hi * 16), lds_ld8(b0p + hi * 16 + 8));
    f16x16 b1 = frag_cat(lds_ld8(b1p + hi * 16), lds_ld8(b1p + hi * 16 + 8));
    c0 = __builtin_amdgcn_wmma_f32_16x16x32_f16(false, a, false, b0, (short)0, c0, false, false);
    c1 = __builtin_amdgcn_wmma_f32_16x16x32_f16(false, a, false, b1, (short)0, c1, false, false);

    wait_async0();
    __syncthreads();
  }

  const int nbase = n0 + tc * 32 + lr;
  const int mrow  = m0 + tr * 16 + hi * 8;
#pragma unroll
  for (int v = 0; v < 8; v++) {
    int m = mrow + v;
    if (m < M) {
      float x0 = c0[v] + bias[nbase];
      float x1 = c1[v] + bias[nbase + 16];
      if (relu) { x0 = fmaxf(x0, 0.f); x1 = fmaxf(x1, 0.f); }
      C[(long)m * ldc + nbase]      = x0;
      C[(long)m * ldc + nbase + 16] = x1;
      if (Ch) {
        Ch[(long)m * ldch + nbase]      = (_Float16)x0;
        Ch[(long)m * ldch + nbase + 16] = (_Float16)x1;
      }
    }
  }
}

// ---------------------------------------------------------------------------
// Flash attention core: O = softmax(Q Kh^T * scale) V, per head.
// grid.x = ceil(n/64), grid.y = heads. block = 128 (4 waves, 16 q-rows each).
// Q/K/V: f16, ld = F, head h occupies columns [h*dk, h*dk+dk). dk in {16,32}.
// K/V tiles double-buffered via async DMA; pad columns kept zero by pre-zeroing.
// ---------------------------------------------------------------------------
__global__ __launch_bounds__(128)
void flash_attn_kernel(const _Float16* __restrict__ Q, int ldq,
                       const _Float16* __restrict__ Km, int ldk,
                       const _Float16* __restrict__ Vm, int ldv,
                       _Float16* __restrict__ O, int ldo,
                       int n, int m, int dk, float scale) {
  __shared__ _Float16 Qs[64][40];         // [q][d]
  __shared__ _Float16 Ks[2][32][40];      // [buf][key][d]
  __shared__ _Float16 Vs[2][32][40];      // [buf][key][d]
  __shared__ _Float16 Ps[4][16][40];      // per-wave probability restage

  const int tid  = threadIdx.x;
  const int wave = tid >> 5;
  const int lane = tid & 31;
  const int lr   = lane & 15;
  const int hi   = lane >> 4;
  const int h    = blockIdx.y;
  const int hoff = h * dk;
  const int q0   = blockIdx.x * 64;

  // pre-zero both K/V buffers (keeps dk=16 padding zero on the async path).
  // NOTE: runtime pointer selection (no constant shared-pointer aggregate;
  // ld.lld cannot relocate addrspacecast in static initializers).
  {
    int which = tid >> 5;        // 0..3
    int row   = tid & 31;
    _Float16* p = (which < 2) ? &Ks[which][row][0] : &Vs[which - 2][row][0];
    f16x8 z = {};
#pragma unroll
    for (int c = 0; c < 40; c += 8) *reinterpret_cast<f16x8*>(p + c) = z;
  }

  // stage Q tile (zero-pad rows>=n and d>=dk)
  {
    int r  = tid >> 1;
    int dh = (tid & 1) * 16;
    bool rowok = (q0 + r) < n;
    const _Float16* qp = Q + (long)(q0 + r) * ldq + hoff;
    *reinterpret_cast<f16x8*>(&Qs[r][dh])     = g_ld8(qp + dh,     rowok && dh < dk);
    *reinterpret_cast<f16x8*>(&Qs[r][dh + 8]) = g_ld8(qp + dh + 8, rowok && (dh + 8) < dk);
  }

  const int j  = tid >> 2;          // key 0..31
  const int dg = (tid & 3) * 8;     // 0,8,16,24
  auto stageKV = [&](int k0, int buf) {
    const _Float16* kp = Km + (long)(k0 + j) * ldk + hoff + dg;
    const _Float16* vp = Vm + (long)(k0 + j) * ldv + hoff + dg;
    if (k0 + 32 <= m) {                 // interior tile: pure DMA
      if (dg < dk) {
        async_b128(&Ks[buf][j][dg], kp);
        async_b128(&Vs[buf][j][dg], vp);
      }
    } else {                            // edge tile: guarded direct stores
      bool ok = (k0 + j) < m && dg < dk;
      *reinterpret_cast<f16x8*>(&Ks[buf][j][dg]) = g_ld8(kp, ok);
      *reinterpret_cast<f16x8*>(&Vs[buf][j][dg]) = g_ld8(vp, ok);
    }
  };

  stageKV(0, 0);
  wait_async0();
  __syncthreads();

  const _Float16* qrow = &Qs[wave * 16 + lr][0];
  f16x16 aQ = frag_cat(lds_ld8(qrow + hi * 16), lds_ld8(qrow + hi * 16 + 8));

  float Mx[8], Ls[8];
#pragma unroll
  for (int v = 0; v < 8; v++) { Mx[v] = -1e30f; Ls[v] = 0.f; }
  f32x8 O0 = {}; f32x8 O1 = {};

  int buf = 0;
  for (int k0 = 0; k0 < m; k0 += 32, buf ^= 1) {
    if (k0 + 32 < m) stageKV(k0 + 32, buf ^ 1);   // DMA next tile under compute

    // scores: B = K^T, b[i] = K[key][hi*16+i] -> contiguous in d
    const _Float16* k0p = &Ks[buf][lr][0];
    const _Float16* k1p = &Ks[buf][16 + lr][0];
    f16x16 bK0 = frag_cat(lds_ld8(k0p + hi * 16), lds_ld8(k0p + hi * 16 + 8));
    f16x16 bK1 = frag_cat(lds_ld8(k1p + hi * 16), lds_ld8(k1p + hi * 16 + 8));
    f32x8 S0 = {}, S1 = {};
    S0 = __builtin_amdgcn_wmma_f32_16x16x32_f16(false, aQ, false, bK0, (short)0, S0, false, false);
    S1 = __builtin_amdgcn_wmma_f32_16x16x32_f16(false, aQ, false, bK1, (short)0, S1, false, false);

    const bool ok0 = (k0 + lr) < m;
    const bool ok1 = (k0 + 16 + lr) < m;
#pragma unroll
    for (int v = 0; v < 8; v++) {
      float s0 = ok0 ? S0[v] * scale : -1e30f;
      float s1 = ok1 ? S1[v] * scale : -1e30f;
      float cm = fmaxf(s0, s1);
#pragma unroll
      for (int off = 1; off < 16; off <<= 1) cm = fmaxf(cm, __shfl_xor(cm, off, 32));
      float Mn = fmaxf(Mx[v], cm);
      float p0 = __expf(s0 - Mn);
      float p1 = __expf(s1 - Mn);
      float rs = p0 + p1;
#pragma unroll
      for (int off = 1; off < 16; off <<= 1) rs += __shfl_xor(rs, off, 32);
      float alpha = __expf(Mx[v] - Mn);
      Ls[v] = Ls[v] * alpha + rs;
      O0[v] *= alpha;
      O1[v] *= alpha;
      Mx[v] = Mn;
      Ps[wave][v + hi * 8][lr]      = (_Float16)p0;   // column scatter (C layout)
      Ps[wave][v + hi * 8][16 + lr] = (_Float16)p1;
    }

    // O += P @ V : A = P (contiguous restage), B = V (K-dim = keys, scalar col reads)
    const _Float16* pp = &Ps[wave][lr][0];
    f16x16 aP = frag_cat(lds_ld8(pp + hi * 16), lds_ld8(pp + hi * 16 + 8));
    f16x16 bV0, bV1;
#pragma unroll
    for (int i = 0; i < 16; i++) {
      bV0[i] = Vs[buf][hi * 16 + i][lr];
      bV1[i] = Vs[buf][hi * 16 + i][16 + lr];
    }
    O0 = __builtin_amdgcn_wmma_f32_16x16x32_f16(false, aP, false, bV0, (short)0, O0, false, false);
    O1 = __builtin_amdgcn_wmma_f32_16x16x32_f16(false, aP, false, bV1, (short)0, O1, false, false);

    wait_async0();
    __syncthreads();
  }

#pragma unroll
  for (int v = 0; v < 8; v++) {
    int row = q0 + wave * 16 + hi * 8 + v;
    if (row < n) {
      float invL = (Ls[v] > 0.f) ? (1.0f / Ls[v]) : 0.f;
      O[(long)row * ldo + hoff + lr] = (_Float16)(O0[v] * invL);
      if (dk > 16) O[(long)row * ldo + hoff + 16 + lr] = (_Float16)(O1[v] * invL);
    }
  }
}

// ---------------------------------------------------------------------------
// Edge-parallel segment sum: prop[row] += val * f[col]  (prop pre-seeded = f)
// ---------------------------------------------------------------------------
__global__ void seg_kernel(const int* __restrict__ row, const int* __restrict__ col,
                           const float* __restrict__ val, const float* __restrict__ f,
                           float* __restrict__ prop, int F, int nE) {
  long idx = (long)blockIdx.x * blockDim.x + threadIdx.x;
  int per = F >> 2;
  long e = idx / per;
  int c4 = (int)(idx % per) * 4;
  if (e < nE) {
    int r = row[e], cNode = col[e];
    float v = val[e];
    float4 fv = *reinterpret_cast<const float4*>(f + (long)cNode * F + c4);
    float* p = prop + (long)r * F + c4;
    atomicAdd(p + 0, v * fv.x);
    atomicAdd(p + 1, v * fv.y);
    atomicAdd(p + 2, v * fv.z);
    atomicAdd(p + 3, v * fv.w);
  }
}

// ---------------------------------------------------------------------------
// cmb16 = f16(prop + features_old) ; features_old read from final columns
// ---------------------------------------------------------------------------
__global__ void combine_kernel(const float* __restrict__ prop, const float* __restrict__ finalPrev,
                               _Float16* __restrict__ cmb, int F, int total) {
  int idx = blockIdx.x * blockDim.x + threadIdx.x;
  if (idx >= total) return;
  int r = idx / F, c = idx % F;
  cmb[idx] = (_Float16)(prop[idx] + finalPrev[(long)r * FINCOLS + c]);
}

// ---------------------------------------------------------------------------
// e16[i] = f16(concat(final[u[i]], final[it[i]+U_NUM]))  (8192 x 896)
// ---------------------------------------------------------------------------
__global__ void gather_kernel(const float* __restrict__ finalBuf, const int* __restrict__ uIdx,
                              const int* __restrict__ iIdx, _Float16* __restrict__ e16, int B) {
  int idx = blockIdx.x * blockDim.x + threadIdx.x;   // B * 224
  if (idx >= B * 224) return;
  int rowi = idx / 224;
  int c4 = (idx % 224) * 4;
  const float* src = (c4 < FINCOLS)
      ? finalBuf + (long)uIdx[rowi] * FINCOLS + c4
      : finalBuf + (long)(iIdx[rowi] + U_NUM) * FINCOLS + (c4 - FINCOLS);
  float4 v = *reinterpret_cast<const float4*>(src);
  _Float16* d = e16 + (long)rowi * 896 + c4;
  d[0] = (_Float16)v.x; d[1] = (_Float16)v.y; d[2] = (_Float16)v.z; d[3] = (_Float16)v.w;
}

// ---------------------------------------------------------------------------
// out[i] = (relu'd h1[i] @ W2 + b2) @ W3 + b3, fp32, one wave per sample
// ---------------------------------------------------------------------------
__global__ void mlp_tail_kernel(const float* __restrict__ h1, const float* __restrict__ W2,
                                const float* __restrict__ b2, const float* __restrict__ W3,
                                const float* __restrict__ b3, float* __restrict__ out, int B) {
  int tid = blockIdx.x * blockDim.x + threadIdx.x;
  int i = tid >> 5;
  int j = tid & 31;
  if (i >= B) return;
  float acc = b2[j];
  const float* hrow = h1 + (long)i * 64;
#pragma unroll
  for (int k = 0; k < 64; k++) acc += hrow[k] * W2[k * 32 + j];
  float part = acc * W3[j];
#pragma unroll
  for (int off = 16; off; off >>= 1) part += __shfl_xor(part, off, 32);
  if (j == 0) out[i] = part + b3[0];
}

// ---------------------------------------------------------------------------
// Host launch
// ---------------------------------------------------------------------------
static inline void launch_gemm(hipStream_t st, const _Float16* A, int lda,
                               const _Float16* BT, int ldbt, const float* bias,
                               float* C, int ldc, _Float16* Ch, int ldch,
                               int M, int N, int K, int relu) {
  dim3 grid((M + 63) / 64, N / 64);
  wmma_gemm_kernel<<<grid, 256, 0, st>>>(A, lda, BT, ldbt, bias, C, ldc, Ch, ldch, M, N, K, relu);
}

extern "C" void kernel_launch(void* const* d_in, const int* in_sizes, int n_in,
                              void* d_out, int out_size, void* d_ws, size_t ws_size,
                              hipStream_t stream) {
  const int Bn = in_sizes[0];                 // 8192
  const int* userIdx = (const int*)d_in[0];
  const int* itemIdx = (const int*)d_in[1];
  const float* uEmbd = (const float*)d_in[2];
  const float* iEmbd = (const float*)d_in[3];
  // ga layers: layer l base index = 4 + 18*l
  //   att_user: qW,qb,kW,kb,vW,vb,oW,ob (+0..+7); att_item (+8..+15); aff W(+16) b(+17)
  const float* t1W = (const float*)d_in[40];
  const float* t1b = (const float*)d_in[41];
  const float* t2W = (const float*)d_in[42];
  const float* t2b = (const float*)d_in[43];
  const float* t3W = (const float*)d_in[44];
  const float* t3b = (const float*)d_in[45];
  const int* L_row = (const int*)d_in[46];
  const int* L_col = (const int*)d_in[47];
  const float* L_val = (const float*)d_in[48];
  const int nEdges = in_sizes[46];

  // ---- carve workspace ----
  char* base = (char*)d_ws;
  size_t cur = 0;
  auto carve = [&](size_t bytes) -> char* {
    char* p = base + cur;
    cur = (cur + bytes + 255) & ~(size_t)255;
    return p;
  };
  float*    finalBuf = (float*)   carve((size_t)N_NODES * FINCOLS * 4);
  _Float16* feat16   = (_Float16*)carve((size_t)N_NODES * EMBED * 2);
  _Float16* qbuf16   = (_Float16*)carve((size_t)N_NODES * EMBED * 2);
  _Float16* kbuf16   = (_Float16*)carve((size_t)N_NODES * EMBED * 2);
  _Float16* vbuf16   = (_Float16*)carve((size_t)N_NODES * EMBED * 2);
  _Float16* aout16   = (_Float16*)carve((size_t)N_NODES * EMBED * 2);
  float*    fbuf32   = (float*)   carve((size_t)N_NODES * EMBED * 4);
  float*    prop32   = (float*)   carve((size_t)N_NODES * EMBED * 4);
  _Float16* cmb16    = (_Float16*)carve((size_t)N_NODES * EMBED * 2);
  _Float16* e16      = (_Float16*)carve((size_t)Bn * 896 * 2);
  float*    h1_32    = (float*)   carve((size_t)Bn * 64 * 4);
  float*    scratch32= (float*)   carve((size_t)N_NODES * EMBED * 4);
  _Float16* w16pool  = (_Float16*)carve((size_t)2 * 1024 * 1024);   // weight f16 pool

  // ---- weight casts (transposed: W[KxN] -> WT[NxK]) ----
  _Float16* w16cur = w16pool;
  auto castWT = [&](const float* src, int K, int N) -> _Float16* {
    _Float16* dst = w16cur;
    long cnt = (long)K * N;
    w16cur += (cnt + 127) & ~(long)127;
    castT_f32_f16<<<dim3((unsigned)((cnt + 255) / 256)), 256, 0, stream>>>(src, dst, K, N);
    return dst;
  };

  const int Fd[2] = {256, 128};
  const int Td[2] = {128, 64};
  const int prevOff[2] = {0, 256};
  const int curOff[2]  = {256, 384};

  _Float16 *Wu[2][4], *Wi[2][4], *Waff[2];
  for (int l = 0; l < 2; l++) {
    int bi = 4 + 18 * l;
    int F = Fd[l], T = Td[l];
    for (int t = 0; t < 4; t++) Wu[l][t] = castWT((const float*)d_in[bi + 2 * t], F, F);
    for (int t = 0; t < 4; t++) Wi[l][t] = castWT((const float*)d_in[bi + 8 + 2 * t], F, F);
    Waff[l] = castWT((const float*)d_in[bi + 16], F, T);
  }
  _Float16* W1T = castWT(t1W, 896, 64);

  // ---- embeddings -> final[:,0:256] and feat16 ----
  {
    long tot = (long)N_NODES * EMBED;
    embed_kernel<<<dim3((unsigned)((tot + 255) / 256)), 256, 0, stream>>>(uEmbd, iEmbd, finalBuf, feat16);
  }

  _Float16* curFeat = feat16;
  int curLd = EMBED;

  for (int l = 0; l < 2; l++) {
    const int F = Fd[l], T = Td[l];
    const int dk = F / 8;
    const float scale = 1.0f / sqrtf((float)dk);
    const int bi = 4 + 18 * l;
    const float* bu[4], * bitem[4];
    for (int t = 0; t < 4; t++) { bu[t] = (const float*)d_in[bi + 2 * t + 1]; bitem[t] = (const float*)d_in[bi + 8 + 2 * t + 1]; }
    const float* affb = (const float*)d_in[bi + 17];

    // --- item_att: q from iF (rows U..), k/v from uF (rows 0..U) ---
    launch_gemm(stream, curFeat + (long)U_NUM * curLd, curLd, Wi[l][0], F, bitem[0],
                scratch32, F, qbuf16, F, I_NUM, F, F, 0);
    launch_gemm(stream, curFeat, curLd, Wi[l][1], F, bitem[1],
                scratch32, F, kbuf16, F, U_NUM, F, F, 0);
    launch_gemm(stream, curFeat, curLd, Wi[l][2], F, bitem[2],
                scratch32, F, vbuf16, F, U_NUM, F, F, 0);
    // --- user_att: q from uF, k/v from iF ---
    launch_gemm(stream, curFeat, curLd, Wu[l][0], F, bu[0],
                scratch32, F, qbuf16 + (long)I_NUM * F, F, U_NUM, F, F, 0);
    launch_gemm(stream, curFeat + (long)U_NUM * curLd, curLd, Wu[l][1], F, bu[1],
                scratch32, F, kbuf16 + (long)U_NUM * F, F, I_NUM, F, F, 0);
    launch_gemm(stream, curFeat + (long)U_NUM * curLd, curLd, Wu[l][2], F, bu[2],
                scratch32, F, vbuf16 + (long)U_NUM * F, F, I_NUM, F, F, 0);

    // --- flash attention (8 heads) ---
    {
      dim3 gi((I_NUM + 63) / 64, 8);
      flash_attn_kernel<<<gi, 128, 0, stream>>>(qbuf16, F, kbuf16, F, vbuf16, F,
                                                aout16, F, I_NUM, U_NUM, dk, scale);
      dim3 gu((U_NUM + 63) / 64, 8);
      flash_attn_kernel<<<gu, 128, 0, stream>>>(qbuf16 + (long)I_NUM * F, F,
                                                kbuf16 + (long)U_NUM * F, F,
                                                vbuf16 + (long)U_NUM * F, F,
                                                aout16 + (long)I_NUM * F, F, U_NUM, I_NUM, dk, scale);
    }

    // --- output projections; f = [item_att ; user_att] (ref's concat order) ---
    launch_gemm(stream, aout16, F, Wi[l][3], F, bitem[3],
                fbuf32, F, ((_Float16*)nullptr), 0, I_NUM, F, F, 0);
    launch_gemm(stream, aout16 + (long)I_NUM * F, F, Wu[l][3], F, bu[3],
                fbuf32 + (long)I_NUM * F, F, ((_Float16*)nullptr), 0, U_NUM, F, F, 0);

    // --- prop = f ; prop += segment_sum(L_val * f[L_col]) ---
    (void)hipMemcpyAsync(prop32, fbuf32, (size_t)N_NODES * F * 4, hipMemcpyDeviceToDevice, stream);
    {
      long work = (long)nEdges * (F / 4);
      seg_kernel<<<dim3((unsigned)((work + 255) / 256)), 256, 0, stream>>>(
          L_row, L_col, L_val, fbuf32, prop32, F, nEdges);
    }

    // --- cmb = prop + features_old (features_old lives in final columns) ---
    {
      int tot = N_NODES * F;
      combine_kernel<<<dim3((tot + 255) / 256), 256, 0, stream>>>(
          prop32, finalBuf + prevOff[l], cmb16, F, tot);
    }

    // --- features = relu(cmb @ Waff + b); write into final columns + feat16 ---
    launch_gemm(stream, cmb16, F, Waff[l], F, affb,
                finalBuf + curOff[l], FINCOLS, feat16, T, N_NODES, T, F, 1);
    curFeat = feat16;
    curLd = T;
  }

  // ---- prediction MLP ----
  gather_kernel<<<dim3((Bn * 224 + 255) / 256), 256, 0, stream>>>(finalBuf, userIdx, itemIdx, e16, Bn);
  launch_gemm(stream, e16, 896, W1T, 896, t1b, h1_32, 64, ((_Float16*)nullptr), 0, Bn, 64, 896, 1);
  mlp_tail_kernel<<<dim3((Bn * 32 + 255) / 256), 256, 0, stream>>>(h1_32, t2W, t2b, t3W, t3b,
                                                                   (float*)d_out, Bn);
}
